// DecoderLayer_56719338111661
// MI455X (gfx1250) — matrix operand order repaired and checked by reading, compiled, not run on hardware
//
#include <hip/hip_runtime.h>
#include <math.h>

typedef __attribute__((ext_vector_type(2)))  float v2f;
typedef __attribute__((ext_vector_type(8)))  float v8f;
typedef __attribute__((ext_vector_type(16))) __bf16 v16bf;
typedef __attribute__((ext_vector_type(4)))  unsigned int u32x4;

#define Bb   2
#define Pp   12
#define Nn   170
#define DMd  64
#define Hh   8
#define DKd  8
#define DHd  32
#define DFd  256
#define POSN 4080        // B*P*N
#define MTILES 255       // POSN/16
#define KMETA 2112       // DH*DM + DM  (outer-product K + bias rows), = 66*32
#define KKSTEPS 66
#define NMETA 192        // 3*H*DK
#define RSQ  0.35355339059327373f   // 1/sqrt(DK)

// ============================================================================
// Pack w2 (+b2 as extra K rows) into WMMA-B-fragment-friendly bf16 blocks:
// Bpk[kk][n][kloc], kloc = k within the 32-wide K step. For k < 2048:
// value = w2[j = k>>6][ n*64 + (k&63) ]; for k >= 2048: value = b2[n*64 + k-2048].
// ============================================================================
__global__ void pack_w2_kernel(const float* __restrict__ w2,
                               const float* __restrict__ b2,
                               __bf16* __restrict__ Bpk) {
  int idx = blockIdx.x * blockDim.x + threadIdx.x;
  if (idx >= KKSTEPS * NMETA * 32) return;
  int kloc = idx & 31;
  int n    = (idx >> 5) % NMETA;
  int kk   = idx / (NMETA * 32);
  int k    = kk * 32 + kloc;
  float v;
  if (k < DHd * DMd) {
    int j = k >> 6, d = k & 63;
    v = w2[(size_t)j * (NMETA * DMd) + n * DMd + d];
  } else {
    v = b2[n * DMd + (k - DHd * DMd)];
  }
  Bpk[idx] = (__bf16)v;
}

// ============================================================================
// Build A matrix rows (vectorized, one b128 store of 8 bf16 per thread):
// Z[pos][k] = rh[pos][k>>6] * x[pos][k&63] for k<2048, else x[pos][k-2048].
// 8 consecutive k share the same rh[j] since 8 | 64.
// ============================================================================
__global__ void build_z_kernel(const float* __restrict__ rh,
                               const float* __restrict__ xin,
                               __bf16* __restrict__ Z) {
  int idx = blockIdx.x * blockDim.x + threadIdx.x;
  if (idx >= POSN * (KMETA / 8)) return;
  int kb  = idx % (KMETA / 8);
  int pos = idx / (KMETA / 8);
  int k = kb * 8;
  union { __bf16 h[8]; u32x4 u; } o;
  if (k < DHd * DMd) {
    float r = rh[pos * DHd + (k >> 6)];
    const float* xp = xin + (size_t)pos * DMd + (k & 63);
#pragma unroll
    for (int t = 0; t < 8; ++t) o.h[t] = (__bf16)(r * xp[t]);
  } else {
    const float* xp = xin + (size_t)pos * DMd + (k - DHd * DMd);
#pragma unroll
    for (int t = 0; t < 8; ++t) o.h[t] = (__bf16)xp[t];
  }
  *(u32x4*)(Z + (size_t)pos * KMETA + k) = o.u;
}

// ============================================================================
// Fused MetaLearner+mhlin GEMM: C[4080,192] = Z[4080,2112] @ Bpk[2112,192]
// bf16 WMMA 16x16x32. Block = one M-tile (4 waves); each wave owns 3 N-tiles
// so each A fragment feeds 3 back-to-back WMMAs (3x A reuse). Software
// pipelined: double-buffered fragment registers so the next k-step's loads
// overlap the current 3-WMMA chain (avoids s_wait_loadcnt 0 drains).
// ============================================================================
typedef union { v16bf v; u32x4 u[2]; } Frag;

__global__ __launch_bounds__(128)
void meta_gemm_kernel(const __bf16* __restrict__ Z,
                      const __bf16* __restrict__ Bpk,
                      float* __restrict__ C) {
  int wave = threadIdx.x >> 5;
  int lane = threadIdx.x & 31;
  int m0 = blockIdx.x * 16;
  int n0 = wave * 48;                        // 3 N-tiles per wave
  int lo = lane & 15, hi = lane >> 4;
  v8f c0 = {0.f, 0.f, 0.f, 0.f, 0.f, 0.f, 0.f, 0.f};
  v8f c1 = c0, c2 = c0;
  const __bf16* Arow = Z + (size_t)(m0 + lo) * KMETA;
  const __bf16* B0 = Bpk + (size_t)(n0 + lo) * 32 + hi * 8;

  auto loadA = [&](Frag& f, int kk) {
    f.u[0] = *(const u32x4*)(Arow + kk * 32 + hi * 8);
    f.u[1] = *(const u32x4*)(Arow + kk * 32 + 16 + hi * 8);
  };
  auto loadB = [&](Frag& f, int kk, int toff) {
    const __bf16* bp = B0 + (size_t)kk * (NMETA * 32) + toff;
    f.u[0] = *(const u32x4*)(bp);
    f.u[1] = *(const u32x4*)(bp + 16);
  };

  Frag aA, p0, p1, p2, aB, q0, q1, q2;
  loadA(aA, 0); loadB(p0, 0, 0); loadB(p1, 0, 512); loadB(p2, 0, 1024);
  for (int kk = 0; kk < KKSTEPS - 2; kk += 2) {
    loadA(aB, kk + 1);
    loadB(q0, kk + 1, 0); loadB(q1, kk + 1, 512); loadB(q2, kk + 1, 1024);
    c0 = __builtin_amdgcn_wmma_f32_16x16x32_bf16(false, aA.v, false, p0.v,
                                                 (short)0, c0, false, false);
    c1 = __builtin_amdgcn_wmma_f32_16x16x32_bf16(false, aA.v, false, p1.v,
                                                 (short)0, c1, false, false);
    c2 = __builtin_amdgcn_wmma_f32_16x16x32_bf16(false, aA.v, false, p2.v,
                                                 (short)0, c2, false, false);
    loadA(aA, kk + 2);
    loadB(p0, kk + 2, 0); loadB(p1, kk + 2, 512); loadB(p2, kk + 2, 1024);
    c0 = __builtin_amdgcn_wmma_f32_16x16x32_bf16(false, aB.v, false, q0.v,
                                                 (short)0, c0, false, false);
    c1 = __builtin_amdgcn_wmma_f32_16x16x32_bf16(false, aB.v, false, q1.v,
                                                 (short)0, c1, false, false);
    c2 = __builtin_amdgcn_wmma_f32_16x16x32_bf16(false, aB.v, false, q2.v,
                                                 (short)0, c2, false, false);
  }
  // epilogue: kk = 64 (in aA/p*), prefetch kk = 65, consume both
  loadA(aB, KKSTEPS - 1);
  loadB(q0, KKSTEPS - 1, 0); loadB(q1, KKSTEPS - 1, 512);
  loadB(q2, KKSTEPS - 1, 1024);
  c0 = __builtin_amdgcn_wmma_f32_16x16x32_bf16(false, aA.v, false, p0.v,
                                               (short)0, c0, false, false);
  c1 = __builtin_amdgcn_wmma_f32_16x16x32_bf16(false, aA.v, false, p1.v,
                                               (short)0, c1, false, false);
  c2 = __builtin_amdgcn_wmma_f32_16x16x32_bf16(false, aA.v, false, p2.v,
                                               (short)0, c2, false, false);
  c0 = __builtin_amdgcn_wmma_f32_16x16x32_bf16(false, aB.v, false, q0.v,
                                               (short)0, c0, false, false);
  c1 = __builtin_amdgcn_wmma_f32_16x16x32_bf16(false, aB.v, false, q1.v,
                                               (short)0, c1, false, false);
  c2 = __builtin_amdgcn_wmma_f32_16x16x32_bf16(false, aB.v, false, q2.v,
                                               (short)0, c2, false, false);

  float* Cp = C + (size_t)n0 + lo;
#pragma unroll
  for (int r = 0; r < 8; ++r) {
    size_t row = (size_t)(m0 + r + 8 * hi) * NMETA;
    Cp[row] = c0[r];
    Cp[row + 16] = c1[r];
    Cp[row + 32] = c2[r];
  }
}

// ============================================================================
// Generic f32 GEMM via v_wmma_f32_16x16x4_f32: C = act(A[M,K]@W[K,N] + bias)
// A-frag (32-bit A 16x4): lane(lo,hi), vgpr v -> k = kk + 2*hi + v.
// B-frag symmetric: n = lo, k = kk + 2*hi + v.
// ============================================================================
__global__ __launch_bounds__(128)
void gemm_f32_kernel(const float* __restrict__ A, const float* __restrict__ W,
                     const float* __restrict__ bias, float* __restrict__ C,
                     int K, int Ncols, int act) {
  int wave = threadIdx.x >> 5;
  int lane = threadIdx.x & 31;
  int nt = blockIdx.y * 4 + wave;
  if (nt * 16 >= Ncols) return;
  int m0 = blockIdx.x * 16, n0 = nt * 16;
  int lo = lane & 15, hi = lane >> 4;
  v8f c = {0.f, 0.f, 0.f, 0.f, 0.f, 0.f, 0.f, 0.f};
  const float* Arow = A + (size_t)(m0 + lo) * K;
  for (int kk = 0; kk < K; kk += 4) {
    v2f a = *(const v2f*)(Arow + kk + 2 * hi);
    const float* wp = W + (size_t)(kk + 2 * hi) * Ncols + n0 + lo;
    v2f b;
    b[0] = wp[0];
    b[1] = wp[Ncols];
    c = __builtin_amdgcn_wmma_f32_16x16x4_f32(false, a, false, b,
                                              (short)0, c, false, false);
  }
  float bv = bias ? bias[n0 + lo] : 0.f;
#pragma unroll
  for (int r = 0; r < 8; ++r) {
    float v = c[r] + bv;
    if (act) v = fmaxf(v, 0.f);
    C[(size_t)(m0 + r + 8 * hi) * Ncols + n0 + lo] = v;
  }
}

// ============================================================================
// RetNet temporal retention: per (b,n,h,q): r[p]=Q.K/sqrt(dk)*D[h,q,p];
// rs=max(|sum r|,1); out[k] = sum_p r[p]/rs * V[p,k].
// qkv layout [pos,192]: Q at +0, K at +64, V at +128, inner = h*8+k.
// ============================================================================
__global__ void retnet_kernel(const float* __restrict__ qkv,
                              const float* __restrict__ Dm,
                              float* __restrict__ out) {
  int t = blockIdx.x * blockDim.x + threadIdx.x;
  if (t >= Bb * Nn * Hh * Pp) return;
  int q = t % Pp;
  int h = (t / Pp) % Hh;
  int n = (t / (Pp * Hh)) % Nn;
  int b = t / (Pp * Hh * Nn);
  float qv[8];
  const float* Qr = qkv + (size_t)((b * Pp + q) * Nn + n) * NMETA + h * 8;
#pragma unroll
  for (int k = 0; k < 8; ++k) qv[k] = Qr[k];
  float r[Pp]; float s = 0.f;
  for (int p = 0; p < Pp; ++p) {
    const float* Kr = qkv + (size_t)((b * Pp + p) * Nn + n) * NMETA + 64 + h * 8;
    float d = 0.f;
#pragma unroll
    for (int k = 0; k < 8; ++k) d += qv[k] * Kr[k];
    d = d * RSQ * Dm[h * (Pp * Pp) + q * Pp + p];
    r[p] = d; s += d;
  }
  float inv = 1.f / fmaxf(fabsf(s), 1.f);
  float acc[8] = {0, 0, 0, 0, 0, 0, 0, 0};
  for (int p = 0; p < Pp; ++p) {
    float w = r[p] * inv;
    const float* Vr = qkv + (size_t)((b * Pp + p) * Nn + n) * NMETA + 128 + h * 8;
#pragma unroll
    for (int k = 0; k < 8; ++k) acc[k] += w * Vr[k];
  }
  float* o = out + (size_t)((b * Pp + q) * Nn + n) * DMd + h * 8;
#pragma unroll
  for (int k = 0; k < 8; ++k) o[k] = acc[k];
}

// ============================================================================
// Temporal encoder-decoder attention (softmax over P) per (b,n,h,q).
// ============================================================================
__global__ void temporal_kernel(const float* __restrict__ qp,
                                const float* __restrict__ kp,
                                const float* __restrict__ vp,
                                float* __restrict__ out) {
  int t = blockIdx.x * blockDim.x + threadIdx.x;
  if (t >= Bb * Nn * Hh * Pp) return;
  int q = t % Pp;
  int h = (t / Pp) % Hh;
  int n = (t / (Pp * Hh)) % Nn;
  int b = t / (Pp * Hh * Nn);
  float qv[8];
  const float* Qr = qp + (size_t)((b * Pp + q) * Nn + n) * DMd + h * 8;
#pragma unroll
  for (int k = 0; k < 8; ++k) qv[k] = Qr[k];
  float s[Pp]; float mx = -3.4e38f;
  for (int p = 0; p < Pp; ++p) {
    const float* Kr = kp + (size_t)((b * Pp + p) * Nn + n) * DMd + h * 8;
    float d = 0.f;
#pragma unroll
    for (int k = 0; k < 8; ++k) d += qv[k] * Kr[k];
    d *= RSQ; s[p] = d; mx = fmaxf(mx, d);
  }
  float den = 0.f;
  for (int p = 0; p < Pp; ++p) { s[p] = expf(s[p] - mx); den += s[p]; }
  float invd = 1.f / den;
  float acc[8] = {0, 0, 0, 0, 0, 0, 0, 0};
  for (int p = 0; p < Pp; ++p) {
    float w = s[p] * invd;
    const float* Vr = vp + (size_t)((b * Pp + p) * Nn + n) * DMd + h * 8;
#pragma unroll
    for (int k = 0; k < 8; ++k) acc[k] += w * Vr[k];
  }
  float* o = out + (size_t)((b * Pp + q) * Nn + n) * DMd + h * 8;
#pragma unroll
  for (int k = 0; k < 8; ++k) o[k] = acc[k];
}

// ============================================================================
// Spatial adjacency-masked attention with nozero-softmax. One block per
// (b,p,h); K/V[170,8] staged in LDS; thread i computes node i's row.
// perBP: 0 => adj is [N,N] shared, 1 => adj is [B,P,N,N].
// ============================================================================
__global__ __launch_bounds__(192)
void spatial_kernel(const float* __restrict__ qkv,
                    const float* __restrict__ adj, int perBP,
                    float* __restrict__ out) {
  int blk = blockIdx.x;
  int h = blk % Hh;
  int p = (blk / Hh) % Pp;
  int b = blk / (Hh * Pp);
  __shared__ float Ks[Nn * DKd];
  __shared__ float Vs[Nn * DKd];
  for (int i = threadIdx.x; i < Nn * DKd; i += blockDim.x) {
    int j = i >> 3, k = i & 7;
    const float* base = qkv + (size_t)((b * Pp + p) * Nn + j) * NMETA + h * 8;
    Ks[i] = base[64 + k];
    Vs[i] = base[128 + k];
  }
  __syncthreads();
  int i = threadIdx.x;
  if (i >= Nn) return;
  int posi = (b * Pp + p) * Nn + i;
  float qv[8];
  const float* Qr = qkv + (size_t)posi * NMETA + h * 8;
#pragma unroll
  for (int k = 0; k < 8; ++k) qv[k] = Qr[k];
  const float* Trow = perBP ? adj + (size_t)posi * Nn : adj + (size_t)i * Nn;
  // pass 1: row max over masked scores (masked entries contribute 0)
  float mx = -3.4e38f;
  for (int j = 0; j < Nn; ++j) {
    float tv = Trow[j];
    float sv = 0.f;
    if (tv != 0.f) {
      float d = 0.f;
#pragma unroll
      for (int k = 0; k < 8; ++k) d += qv[k] * Ks[j * 8 + k];
      sv = d * RSQ;
    }
    mx = fmaxf(mx, sv);
  }
  // pass 2: e = exp(s-mx)*mask; alpha = e/(sum e + 1e-5) * T
  float den = 0.f;
  float acc[8] = {0, 0, 0, 0, 0, 0, 0, 0};
  for (int j = 0; j < Nn; ++j) {
    float tv = Trow[j];
    if (tv != 0.f) {
      float d = 0.f;
#pragma unroll
      for (int k = 0; k < 8; ++k) d += qv[k] * Ks[j * 8 + k];
      d *= RSQ;
      if (d != 0.f) {                // mask = (s*(T!=0) != 0)
        float e = expf(d - mx);
        den += e;
        float w = e * tv;
#pragma unroll
        for (int k = 0; k < 8; ++k) acc[k] += w * Vs[j * 8 + k];
      }
    }
  }
  float invd = 1.f / (den + 1e-5f);
  float* o = out + (size_t)posi * DMd + h * 8;
#pragma unroll
  for (int k = 0; k < 8; ++k) o[k] = acc[k] * invd;
}

// ============================================================================
// Gated dynamic connection, G=8 heads (C=8). One thread per (pos,d).
// ============================================================================
__global__ void gdc8_kernel(const float* __restrict__ att,
                            const float* __restrict__ W1,
                            const float* __restrict__ W2,
                            float* __restrict__ out) {
  int t = blockIdx.x * blockDim.x + threadIdx.x;
  if (t >= POSN * DMd) return;
  int d = t & 63, pos = t >> 6;
  const float* a = att + (size_t)pos * DMd;
  float av[8], sv[8];
  float smax = -3.4e38f;
#pragma unroll
  for (int g = 0; g < 8; ++g) {
    float aa = 0.f, ss = 0.f;
#pragma unroll
    for (int c = 0; c < 8; ++c) {
      float x = a[g * 8 + c];
      aa += x * W1[(g * 8 + c) * DMd + d];
      ss += x * W2[(g * 8 + c) * DMd + d];
    }
    ss = fmaxf(ss, 0.f);
    av[g] = aa; sv[g] = ss;
    smax = fmaxf(smax, ss);
  }
  float den = 0.f, o = 0.f;
#pragma unroll
  for (int g = 0; g < 8; ++g) {
    float e = expf(sv[g] - smax);
    den += e; o += av[g] * e;
  }
  out[t] = o / den;
}

// ============================================================================
// Gated dynamic connection, G=2 (C=64). One thread per (pos,d).
// ============================================================================
__global__ void gdc2_kernel(const float* __restrict__ o0,
                            const float* __restrict__ o1,
                            const float* __restrict__ W1,
                            const float* __restrict__ W2,
                            float* __restrict__ out) {
  int t = blockIdx.x * blockDim.x + threadIdx.x;
  if (t >= POSN * DMd) return;
  int d = t & 63, pos = t >> 6;
  const float* r0 = o0 + (size_t)pos * DMd;
  const float* r1 = o1 + (size_t)pos * DMd;
  float a0 = 0.f, a1 = 0.f, s0 = 0.f, s1 = 0.f;
  for (int c = 0; c < DMd; ++c) {
    float x0 = r0[c], x1 = r1[c];
    a0 += x0 * W1[c * DMd + d];
    s0 += x0 * W2[c * DMd + d];
    a1 += x1 * W1[(DMd + c) * DMd + d];
    s1 += x1 * W2[(DMd + c) * DMd + d];
  }
  s0 = fmaxf(s0, 0.f); s1 = fmaxf(s1, 0.f);
  float m = fmaxf(s0, s1);
  float e0 = expf(s0 - m), e1 = expf(s1 - m);
  out[t] = (a0 * e0 + a1 * e1) / (e0 + e1);
}

// ============================================================================
// Swish gating elementwise: h = t1*o; u = h*sigmoid(h)
// ============================================================================
__global__ void gate_kernel(const float* __restrict__ t1,
                            const float* __restrict__ o,
                            float* __restrict__ u) {
  int i = blockIdx.x * blockDim.x + threadIdx.x;
  if (i >= POSN * DMd) return;
  float h = t1[i] * o[i];
  u[i] = h / (1.f + expf(-h));
}

// ============================================================================
// LayerNorm(y + resid) over DM=64, one thread per row.
// ============================================================================
__global__ void add_ln_kernel(const float* __restrict__ y,
                              const float* __restrict__ resid,
                              const float* __restrict__ g,
                              const float* __restrict__ be,
                              float* __restrict__ out) {
  int pos = blockIdx.x * blockDim.x + threadIdx.x;
  if (pos >= POSN) return;
  float buf[DMd];
  float m = 0.f;
#pragma unroll
  for (int d = 0; d < DMd; ++d) {
    float v = y[(size_t)pos * DMd + d] + resid[(size_t)pos * DMd + d];
    buf[d] = v; m += v;
  }
  m *= (1.f / DMd);
  float var = 0.f;
#pragma unroll
  for (int d = 0; d < DMd; ++d) { float t = buf[d] - m; var += t * t; }
  var *= (1.f / DMd);
  float inv = rsqrtf(var + 1e-5f);
#pragma unroll
  for (int d = 0; d < DMd; ++d)
    out[(size_t)pos * DMd + d] = (buf[d] - m) * inv * g[d] + be[d];
}

// ============================================================================
// Host launcher
// ============================================================================
extern "C" void kernel_launch(void* const* d_in, const int* in_sizes, int n_in,
                              void* d_out, int out_size, void* d_ws, size_t ws_size,
                              hipStream_t stream) {
  const float* x    = (const float*)d_in[0];
  const float* c_x  = (const float*)d_in[1];
  const float* enc  = (const float*)d_in[2];
  const float* Tm   = (const float*)d_in[3];
  const float* Am   = (const float*)d_in[4];
  const float* Dm   = (const float*)d_in[5];
  const float* mr_w1 = (const float*)d_in[6];  const float* mr_b1 = (const float*)d_in[7];
  const float* mr_w2 = (const float*)d_in[8];  const float* mr_b2 = (const float*)d_in[9];
  const float* ms0_w1 = (const float*)d_in[10]; const float* ms0_b1 = (const float*)d_in[11];
  const float* ms0_w2 = (const float*)d_in[12]; const float* ms0_b2 = (const float*)d_in[13];
  const float* ms1_w1 = (const float*)d_in[14]; const float* ms1_b1 = (const float*)d_in[15];
  const float* ms1_w2 = (const float*)d_in[16]; const float* ms1_b2 = (const float*)d_in[17];
  const float* gr_W1 = (const float*)d_in[18]; const float* gr_W2 = (const float*)d_in[19];
  const float* gs0_W1 = (const float*)d_in[20]; const float* gs0_W2 = (const float*)d_in[21];
  const float* gs1_W1 = (const float*)d_in[22]; const float* gs1_W2 = (const float*)d_in[23];
  const float* ge_W1 = (const float*)d_in[24]; const float* ge_W2 = (const float*)d_in[25];
  const float* g2_W1 = (const float*)d_in[26]; const float* g2_W2 = (const float*)d_in[27];
  const float* swr_wg = (const float*)d_in[28]; const float* swr_bg = (const float*)d_in[29];
  const float* swr_wo = (const float*)d_in[30]; const float* swr_bo = (const float*)d_in[31];
  const float* sws_wg = (const float*)d_in[32]; const float* sws_bg = (const float*)d_in[33];
  const float* sws_wo = (const float*)d_in[34]; const float* sws_bo = (const float*)d_in[35];
  const float* swe_wg = (const float*)d_in[36]; const float* swe_bg = (const float*)d_in[37];
  const float* swe_wo = (const float*)d_in[38]; const float* swe_bo = (const float*)d_in[39];
  const float* lnr_g = (const float*)d_in[40]; const float* lnr_b = (const float*)d_in[41];
  const float* lns_g = (const float*)d_in[42]; const float* lns_b = (const float*)d_in[43];
  const float* lne_g = (const float*)d_in[44]; const float* lne_b = (const float*)d_in[45];
  const float* lnf_g = (const float*)d_in[46]; const float* lnf_b = (const float*)d_in[47];
  const float* wq = (const float*)d_in[48];
  const float* wk = (const float*)d_in[49];
  const float* wv = (const float*)d_in[50];
  const float* f_w1 = (const float*)d_in[51]; const float* f_b1 = (const float*)d_in[52];
  const float* f_w2 = (const float*)d_in[53]; const float* f_b2 = (const float*)d_in[54];

  // ---- workspace bump allocator ----
  char* cur = (char*)d_ws;
  auto alloc = [&](size_t n) -> void* {
    void* p = (void*)cur;
    cur += (n + 255) & ~(size_t)255;
    return p;
  };
  __bf16* bpk0 = (__bf16*)alloc(KKSTEPS * NMETA * 32 * sizeof(__bf16));
  __bf16* bpk1 = (__bf16*)alloc(KKSTEPS * NMETA * 32 * sizeof(__bf16));
  __bf16* bpk2 = (__bf16*)alloc(KKSTEPS * NMETA * 32 * sizeof(__bf16));
  __bf16* Z    = (__bf16*)alloc((size_t)POSN * KMETA * sizeof(__bf16));
  float* rh0 = (float*)alloc((size_t)POSN * DHd * 4);
  float* rh1 = (float*)alloc((size_t)POSN * DHd * 4);
  float* rh2 = (float*)alloc((size_t)POSN * DHd * 4);
  float* qkv0 = (float*)alloc((size_t)POSN * NMETA * 4);
  float* qkv1 = (float*)alloc((size_t)POSN * NMETA * 4);
  float* qkv2 = (float*)alloc((size_t)POSN * NMETA * 4);
  size_t rowb = (size_t)POSN * DMd * 4;
  float* attR = (float*)alloc(rowb);
  float* oR   = (float*)alloc(rowb);
  float* t1   = (float*)alloc(rowb);
  float* u    = (float*)alloc(rowb);
  float* y    = (float*)alloc(rowb);
  float* x1   = (float*)alloc(rowb);
  float* att0 = (float*)alloc(rowb);
  float* att1 = (float*)alloc(rowb);
  float* o0   = (float*)alloc(rowb);
  float* o1   = (float*)alloc(rowb);
  float* og   = (float*)alloc(rowb);
  float* x2   = (float*)alloc(rowb);
  float* qp   = (float*)alloc(rowb);
  float* kp   = (float*)alloc(rowb);
  float* vp   = (float*)alloc(rowb);
  float* attE = (float*)alloc(rowb);
  float* oE   = (float*)alloc(rowb);
  float* x3   = (float*)alloc(rowb);
  float* ffh  = (float*)alloc((size_t)POSN * DFd * 4);

  auto gemm = [&](const float* A, const float* W, const float* bias, float* C,
                  int K, int Ncols, int act) {
    dim3 g(MTILES, (Ncols / 16 + 3) / 4);
    gemm_f32_kernel<<<g, 128, 0, stream>>>(A, W, bias, C, K, Ncols, act);
  };

  // ---- pack metalearner B matrices (bf16) ----
  {
    int tot = KKSTEPS * NMETA * 32;
    int nb = (tot + 255) / 256;
    pack_w2_kernel<<<nb, 256, 0, stream>>>(mr_w2,  mr_b2,  bpk0);
    pack_w2_kernel<<<nb, 256, 0, stream>>>(ms0_w2, ms0_b2, bpk1);
    pack_w2_kernel<<<nb, 256, 0, stream>>>(ms1_w2, ms1_b2, bpk2);
  }
  // rh = relu(c_x @ w1 + b1) for all three metalearners (c_x shared)
  gemm(c_x, mr_w1,  mr_b1,  rh0, DMd, DHd, 1);
  gemm(c_x, ms0_w1, ms0_b1, rh1, DMd, DHd, 1);
  gemm(c_x, ms1_w1, ms1_b1, rh2, DMd, DHd, 1);

  int zblocks = (POSN * (KMETA / 8) + 255) / 256;

  // ===== 1) RetNet temporal retention =====
  build_z_kernel<<<zblocks, 256, 0, stream>>>(rh0, x, Z);
  meta_gemm_kernel<<<MTILES, 128, 0, stream>>>(Z, bpk0, qkv0);
  retnet_kernel<<<128, 256, 0, stream>>>(qkv0, Dm, attR);
  gdc8_kernel<<<(POSN * DMd) / 256, 256, 0, stream>>>(attR, gr_W1, gr_W2, oR);
  gemm(x, swr_wg, swr_bg, t1, DMd, DMd, 0);
  gate_kernel<<<(POSN * DMd) / 256, 256, 0, stream>>>(t1, oR, u);
  gemm(u, swr_wo, swr_bo, y, DMd, DMd, 0);
  add_ln_kernel<<<(POSN + 127) / 128, 128, 0, stream>>>(y, x, lnr_g, lnr_b, x1);

  // ===== 2) Spatial attention (predefined T + adaptive A) =====
  build_z_kernel<<<zblocks, 256, 0, stream>>>(rh1, x1, Z);
  meta_gemm_kernel<<<MTILES, 128, 0, stream>>>(Z, bpk1, qkv1);
  build_z_kernel<<<zblocks, 256, 0, stream>>>(rh2, x1, Z);
  meta_gemm_kernel<<<MTILES, 128, 0, stream>>>(Z, bpk2, qkv2);
  spatial_kernel<<<Bb * Pp * Hh, 192, 0, stream>>>(qkv1, Tm, 0, att0);
  spatial_kernel<<<Bb * Pp * Hh, 192, 0, stream>>>(qkv2, Am, 1, att1);
  gdc8_kernel<<<(POSN * DMd) / 256, 256, 0, stream>>>(att0, gs0_W1, gs0_W2, o0);
  gdc8_kernel<<<(POSN * DMd) / 256, 256, 0, stream>>>(att1, gs1_W1, gs1_W2, o1);
  gdc2_kernel<<<(POSN * DMd) / 256, 256, 0, stream>>>(o0, o1, g2_W1, g2_W2, og);
  gemm(x1, sws_wg, sws_bg, t1, DMd, DMd, 0);
  gate_kernel<<<(POSN * DMd) / 256, 256, 0, stream>>>(t1, og, u);
  gemm(u, sws_wo, sws_bo, y, DMd, DMd, 0);
  add_ln_kernel<<<(POSN + 127) / 128, 128, 0, stream>>>(y, x1, lns_g, lns_b, x2);

  // ===== 3) Temporal encoder-decoder attention =====
  gemm(x2,  wq, nullptr, qp, DMd, DMd, 0);
  gemm(enc, wk, nullptr, kp, DMd, DMd, 0);
  gemm(enc, wv, nullptr, vp, DMd, DMd, 0);
  temporal_kernel<<<128, 256, 0, stream>>>(qp, kp, vp, attE);
  gdc8_kernel<<<(POSN * DMd) / 256, 256, 0, stream>>>(attE, ge_W1, ge_W2, oE);
  gemm(x2, swe_wg, swe_bg, t1, DMd, DMd, 0);
  gate_kernel<<<(POSN * DMd) / 256, 256, 0, stream>>>(t1, oE, u);
  gemm(u, swe_wo, swe_bo, y, DMd, DMd, 0);
  add_ln_kernel<<<(POSN + 127) / 128, 128, 0, stream>>>(y, x2, lne_g, lne_b, x3);

  // ===== 4) FFN =====
  gemm(x3, f_w1, f_b1, ffh, DMd, DFd, 1);
  gemm(ffh, f_w2, f_b2, y, DFd, DMd, 0);
  add_ln_kernel<<<(POSN + 127) / 128, 128, 0, stream>>>(y, x3, lnf_g, lnf_b,
                                                        (float*)d_out);
  (void)in_sizes; (void)n_in; (void)out_size; (void)ws_size;
}
